// SelfNavigatedMambaBlock_21818433863807
// MI455X (gfx1250) — matrix operand order repaired and checked by reading, not compiled, not run on hardware
//
#include <hip/hip_runtime.h>
#include <hip/hip_bf16.h>

// ---------------------------------------------------------------------------
// SelfNavigatedMambaBlock for MI455X (gfx1250, wave32, WMMA f32<=f16 16x16x32)
//   B=4, L=4096, C=512, d_inner=1024, d_state=4, dt_rank=32, d_conv=4
// All three heavy GEMMs (3x3 conv implicit GEMM, in_proj, out_proj) use
// v_wmma_f32_16x16x32_f16 with the shared 64xK weight strip staged through
// LDS by GLOBAL_LOAD_ASYNC_TO_LDS_B128 (hand-unrolled double buffer,
// ASYNCcnt + workgroup-barrier synchronized). A-fragments are pure
// global_load_b128 (single base + row index; OOB conv taps hit a zeroed row).
// Selective scan uses a 3-pass chunked associative scan (16x parallelism).
// ---------------------------------------------------------------------------

typedef __attribute__((ext_vector_type(16))) _Float16 v16h;
typedef __attribute__((ext_vector_type(8)))  _Float16 v8h;
typedef __attribute__((ext_vector_type(8)))  float    v8f;
typedef __attribute__((ext_vector_type(4)))  int      v4i;

union V16 { struct { v8h lo, hi; } p; v16h v; };

#define BB 4
#define LL 4096
#define CC 512
#define DI 1024
#define BL (BB*LL)      // 16384
#define BLC ((size_t)BL*CC)
#define BLD ((size_t)BL*DI)
#define NCH 16          // scan chunks
#define CHL 256         // chunk length

#if defined(__AMDGCN__) && __has_builtin(__builtin_amdgcn_global_load_async_to_lds_b128)
#define HAVE_ASYNC_LDS 1
#else
#define HAVE_ASYNC_LDS 0
#endif

#define AS1 __attribute__((address_space(1)))
#define AS3 __attribute__((address_space(3)))

// one 16B chunk, per-lane: global -> LDS (async on CDNA5, tracked by ASYNCcnt)
__device__ __forceinline__ void async_cp16(const _Float16* g, _Float16* l) {
#if HAVE_ASYNC_LDS
  __builtin_amdgcn_global_load_async_to_lds_b128(
      (AS1 v4i*)g, (AS3 v4i*)l, 0, 0);
#else
  *(v8h*)l = *(const v8h*)g;
#endif
}

__device__ __forceinline__ void wait_async() {
#if HAVE_ASYNC_LDS
#if __has_builtin(__builtin_amdgcn_s_wait_asynccnt)
  __builtin_amdgcn_s_wait_asynccnt((short)0);
#else
  asm volatile("s_wait_asynccnt 0x0" ::: "memory");
#endif
#endif
}

// A fragment: 16x32 f16, M = lane&15.
// lane<16 : halves 0..7 = K k0..k0+7,  halves 8..15 = K k0+16..k0+23
// lane>=16: halves 0..7 = K k0+8..+15, halves 8..15 = K k0+24..k0+31
__device__ __forceinline__ v16h load_a16(const _Float16* rowp, int k0, int lane) {
  int kb = k0 + ((lane & 16) ? 8 : 0);
  V16 u;
  u.p.lo = *(const v8h*)(rowp + kb);
  u.p.hi = *(const v8h*)(rowp + kb + 16);
  return u.v;
}

// B fragment read from the LDS-staged strip (row stride BSTRIDE halfs).
// lane&15 selects N row of W; lane>=16 takes K k0+8..+15 and k0+24..+31 (lane<16: k0..+7, k0+16..+23).
#define BSTRIDE 40      // 80B = 20 dwords: bank-conflict-free b128 reads
__device__ __forceinline__ v16h load_b_lds(const _Float16* strip, int j, int lane) {
  const _Float16* brow = strip + (j * 16 + (lane & 15)) * BSTRIDE + ((lane & 16) ? 8 : 0);
  V16 u;
  u.p.lo = *(const v8h*)brow;
  u.p.hi = *(const v8h*)(brow + 16);
  return u.v;
}

// 4 accumulating WMMAs against one A fragment and a 64-row LDS strip
__device__ __forceinline__ void wmma4(v8f acc[4], v16h a, const _Float16* strip, int lane) {
#pragma unroll
  for (int j = 0; j < 4; ++j)
    acc[j] = __builtin_amdgcn_wmma_f32_16x16x32_f16(
        false, a, false, load_b_lds(strip, j, lane), (short)0, acc[j], false, false);
}

// ---------------- stage 1: residual add + RMSNorm --------------------------
__global__ __launch_bounds__(256) void norm_kernel(
    const float* __restrict__ hs, const float* __restrict__ resid,
    const float* __restrict__ w, float* __restrict__ res_out,
    _Float16* __restrict__ nrm16) {
  int row = blockIdx.x;                       // b*L + l
  const float* hp = hs    + (size_t)row * CC;
  const float* rp = resid + (size_t)row * CC;
  int t = threadIdx.x;
  float v0 = hp[t] + rp[t];
  float v1 = hp[t + 256] + rp[t + 256];
  __shared__ float sred[256];
  sred[t] = v0 * v0 + v1 * v1;
  __syncthreads();
  for (int s = 128; s > 0; s >>= 1) {
    if (t < s) sred[t] += sred[t + s];
    __syncthreads();
  }
  float inv = rsqrtf(sred[0] * (1.0f / CC) + 1e-5f);
  res_out[(size_t)row * CC + t]       = v0;
  res_out[(size_t)row * CC + t + 256] = v1;
  nrm16[(size_t)row * CC + t]         = (_Float16)(v0 * inv * w[t]);
  nrm16[(size_t)row * CC + t + 256]   = (_Float16)(v1 * inv * w[t + 256]);
}

// ---------------- weight conversion / repack -------------------------------
__global__ void cvt_f16_kernel(const float* __restrict__ s, _Float16* __restrict__ d, int n) {
  int i = blockIdx.x * blockDim.x + threadIdx.x;
  if (i < n) d[i] = (_Float16)s[i];
}

__global__ void zero_f16_kernel(_Float16* __restrict__ p, int n) {
  int i = blockIdx.x * blockDim.x + threadIdx.x;
  if (i < n) p[i] = (_Float16)0.0f;
}

// conv_w (Co,Ci,3,3) -> w9[t][co][ci] f16, t = ky*3+kx
__global__ void repack_conv_w(const float* __restrict__ w, _Float16* __restrict__ w9) {
  int tid = blockIdx.x * blockDim.x + threadIdx.x;        // 9*512*512
  if (tid >= 9 * CC * CC) return;
  int ci = tid & 511, co = (tid >> 9) & 511, t = tid >> 18;
  w9[tid] = (_Float16)w[((size_t)co * CC + ci) * 9 + t];
}

// ---------------- stable partition of mask (zeros first) -------------------
__global__ void build_idx(const int* __restrict__ mask, int* __restrict__ idx) {
  int b = threadIdx.x;
  if (b >= BB) return;
  int cnt = 0;
  for (int l = 0; l < LL; ++l) if (mask[b * LL + l] == 0) idx[b * LL + cnt++] = l;
  for (int l = 0; l < LL; ++l) if (mask[b * LL + l] != 0) idx[b * LL + cnt++] = l;
}

// ---------------- 3x3 conv as implicit GEMM (WMMA, async-LDS B) ------------
// A = nrm16 (BL+1 NHWC rows, row BL zeroed), strip kk = w9[kk>>4] k-chunk.
// Workgroup 128M x 64N. OOB taps resolve to row BL -> branchless global
// loads, EXEC stays all-1s, no flat path, no pointer arrays.
__device__ __forceinline__ int conv_row(int b, int y, int xr, int t) {
  int dy = t / 3 - 1, dx = t % 3 - 1;
  int yy = y + dy, xx = xr + dx;
  bool ok = (yy >= 0) & (yy < 64) & (xx >= 0) & (xx < 64);
  return ok ? ((b << 12) + (yy << 6) + xx) : BL;
}

__global__ __launch_bounds__(256) void conv3x3_wmma(
    const _Float16* __restrict__ A, const _Float16* __restrict__ W9,
    const float* __restrict__ bias, _Float16* __restrict__ O16) {
  const int K = CC, N = CC;
  __shared__ _Float16 sB0[64 * BSTRIDE], sB1[64 * BSTRIDE];
  int tid = threadIdx.x, wv = tid >> 5, lane = tid & 31;
  int mt = blockIdx.x >> 3, nt = blockIdx.x & 7;    // 128 x 8 blocks
  int m0 = mt * 128 + wv * 16, n0 = nt * 64;
  int b  = m0 >> 12;
  int l0 = m0 & 4095;
  int y  = l0 >> 6;
  int xr = (l0 & 63) + (lane & 15);                 // this lane's A-row x coord
  int srow = tid >> 2, sc = tid & 3;
  const _Float16* gW = W9 + (size_t)(n0 + srow) * K + sc * 8;
  _Float16* ld0 = &sB0[srow * BSTRIDE + sc * 8];
  _Float16* ld1 = &sB1[srow * BSTRIDE + sc * 8];

  async_cp16(gW, ld0);                              // strip kk = 0
  wait_async();
  __syncthreads();

  v8f acc[4] = {{}, {}, {}, {}};
  for (int kk = 0; kk < 144; kk += 2) {             // 9 taps x 16 k-chunks
    // phase 0: prefetch kk+1 -> sB1, compute kk from sB0
    async_cp16(gW + (size_t)((kk + 1) >> 4) * N * K + ((kk + 1) & 15) * 32, ld1);
    {
      int t = kk >> 4, k0 = (kk & 15) * 32;
      v16h a = load_a16(A + (size_t)conv_row(b, y, xr, t) * K, k0, lane);
      wmma4(acc, a, sB0, lane);
    }
    wait_async();
    __syncthreads();
    // phase 1: prefetch kk+2 -> sB0, compute kk+1 from sB1
    if (kk + 2 < 144)
      async_cp16(gW + (size_t)((kk + 2) >> 4) * N * K + ((kk + 2) & 15) * 32, ld0);
    {
      int t = (kk + 1) >> 4, k0 = ((kk + 1) & 15) * 32;
      v16h a = load_a16(A + (size_t)conv_row(b, y, xr, t) * K, k0, lane);
      wmma4(acc, a, sB1, lane);
    }
    wait_async();
    __syncthreads();
  }
  int rbase = m0 + ((lane & 16) ? 8 : 0);
  int col0  = lane & 15;
#pragma unroll
  for (int j = 0; j < 4; ++j) {
    float bv = bias[n0 + j * 16 + col0];
#pragma unroll
    for (int v = 0; v < 8; ++v)
      O16[(size_t)(rbase + v) * N + n0 + j * 16 + col0] = (_Float16)(acc[j][v] + bv);
  }
}

// ---------------- gather rows into "easy" order ----------------------------
__global__ void gather_rows(const _Float16* __restrict__ src, const int* __restrict__ idx,
                            _Float16* __restrict__ dst) {
  size_t tid = (size_t)blockIdx.x * blockDim.x + threadIdx.x;   // BL * 64 chunks
  if (tid >= BLC / 8) return;
  int row = (int)(tid >> 6), c8 = (int)(tid & 63);
  int b = row >> 12, i = row & 4095;
  int l = idx[b * LL + i];
  ((v8h*)dst)[(size_t)row * 64 + c8] = ((const v8h*)src)[((size_t)(b << 12) + l) * 64 + c8];
}

// ---------------- WMMA GEMM, B staged via async-LDS double buffer ----------
// C[m][n] = sum_k A[m][k]*W[n][k]. Workgroup = 128M x 64N (8 waves x 16M).
// Requires M % 128 == 0, N % 64 == 0, K % 64 == 0.
__global__ __launch_bounds__(256) void gemm_f16_lds(
    const _Float16* __restrict__ A, const _Float16* __restrict__ W,
    float* __restrict__ C, int M, int N, int K) {
  __shared__ _Float16 sB0[64 * BSTRIDE], sB1[64 * BSTRIDE];
  int tid = threadIdx.x, wv = tid >> 5, lane = tid & 31;
  int nTiles = N >> 6;
  int mt = blockIdx.x / nTiles, nt = blockIdx.x % nTiles;
  int m0 = mt * 128 + wv * 16, n0 = nt * 64;
  int srow = tid >> 2, sc = tid & 3;
  const _Float16* gsrc = W + (size_t)(n0 + srow) * K + sc * 8;
  _Float16* ld0 = &sB0[srow * BSTRIDE + sc * 8];
  _Float16* ld1 = &sB1[srow * BSTRIDE + sc * 8];

  async_cp16(gsrc, ld0);                   // preload k0 = 0
  wait_async();
  __syncthreads();

  v8f acc[4] = {{}, {}, {}, {}};
  const _Float16* arow = A + (size_t)(m0 + (lane & 15)) * K;
  for (int k0 = 0; k0 < K; k0 += 64) {
    // phase 0: prefetch k0+32 -> sB1, compute k0 from sB0
    async_cp16(gsrc + k0 + 32, ld1);
    wmma4(acc, load_a16(arow, k0, lane), sB0, lane);
    wait_async();
    __syncthreads();
    // phase 1: prefetch k0+64 -> sB0, compute k0+32 from sB1
    if (k0 + 64 < K) async_cp16(gsrc + k0 + 64, ld0);
    wmma4(acc, load_a16(arow, k0 + 32, lane), sB1, lane);
    wait_async();
    __syncthreads();
  }
  int rbase = m0 + ((lane & 16) ? 8 : 0);
  int col0  = lane & 15;
#pragma unroll
  for (int j = 0; j < 4; ++j)
#pragma unroll
    for (int v = 0; v < 8; ++v)
      C[(size_t)(rbase + v) * N + n0 + j * 16 + col0] = acc[j][v];
}

// ---------------- out_proj GEMM (async-LDS B) + scatter --------------------
__global__ __launch_bounds__(256) void gemm_out_lds(
    const _Float16* __restrict__ Y16, const _Float16* __restrict__ W,
    const int* __restrict__ idx, float* __restrict__ out) {
  const int N = CC, K = DI;
  __shared__ _Float16 sB0[64 * BSTRIDE], sB1[64 * BSTRIDE];
  int tid = threadIdx.x, wv = tid >> 5, lane = tid & 31;
  int mt = blockIdx.x >> 3, nt = blockIdx.x & 7;
  int m0 = mt * 128 + wv * 16, n0 = nt * 64;
  int srow = tid >> 2, sc = tid & 3;
  const _Float16* gsrc = W + (size_t)(n0 + srow) * K + sc * 8;
  _Float16* ld0 = &sB0[srow * BSTRIDE + sc * 8];
  _Float16* ld1 = &sB1[srow * BSTRIDE + sc * 8];

  async_cp16(gsrc, ld0);
  wait_async();
  __syncthreads();

  v8f acc[4] = {{}, {}, {}, {}};
  const _Float16* arow = Y16 + (size_t)(m0 + (lane & 15)) * K;
  for (int k0 = 0; k0 < K; k0 += 64) {
    async_cp16(gsrc + k0 + 32, ld1);
    wmma4(acc, load_a16(arow, k0, lane), sB0, lane);
    wait_async();
    __syncthreads();
    if (k0 + 64 < K) async_cp16(gsrc + k0 + 64, ld0);
    wmma4(acc, load_a16(arow, k0 + 32, lane), sB1, lane);
    wait_async();
    __syncthreads();
  }
  int rbase = m0 + ((lane & 16) ? 8 : 0);
  int col0  = lane & 15;
  int ldest[8];
#pragma unroll
  for (int v = 0; v < 8; ++v) {
    int row = rbase + v;
    int b = row >> 12, i = row & 4095;
    ldest[v] = (b << 12) + idx[b * LL + i];
  }
#pragma unroll
  for (int j = 0; j < 4; ++j)
#pragma unroll
    for (int v = 0; v < 8; ++v)
      out[(size_t)ldest[v] * N + n0 + j * 16 + col0] = acc[j][v];
}

// ---------------- depthwise causal conv1d + SiLU ---------------------------
__global__ void dwconv1d_silu(const float* __restrict__ xz, const float* __restrict__ w,
                              const float* __restrict__ bias, float* __restrict__ xc) {
  size_t tid = (size_t)blockIdx.x * blockDim.x + threadIdx.x;   // BL * DI
  if (tid >= BLD) return;
  int d = (int)(tid & (DI - 1));
  int bl = (int)(tid >> 10);
  int b = bl >> 12, l = bl & 4095;
  float acc = bias[d];
#pragma unroll
  for (int j = 0; j < 4; ++j) {
    int ls = l - 3 + j;
    if (ls >= 0) acc += w[d * 4 + j] * xz[((size_t)((b << 12) + ls)) * 2048 + d];
  }
  xc[tid] = acc / (1.0f + __expf(-acc));
}

// ---------------- x_proj: x_dbl[bl][e] = sum_k xc[bl][k]*W[e][k], e<40 -----
__global__ void xproj_kernel(const float* __restrict__ xc, const float* __restrict__ W,
                             float* __restrict__ xdbl) {
  int tid = blockIdx.x * blockDim.x + threadIdx.x;              // BL * 40
  if (tid >= BL * 40) return;
  int e = tid % 40, bl = tid / 40;
  const float* row = xc + (size_t)bl * DI;
  const float* wr  = W + (size_t)e * DI;
  float acc = 0.f;
  for (int k = 0; k < DI; ++k) acc += row[k] * wr[k];
  xdbl[tid] = acc;
}

// ---------------- dt = softplus(dtr @ dtW^T + b) ---------------------------
__global__ void dtproj_kernel(const float* __restrict__ xdbl, const float* __restrict__ W,
                              const float* __restrict__ bias, float* __restrict__ dt) {
  size_t tid = (size_t)blockIdx.x * blockDim.x + threadIdx.x;   // BL * DI
  if (tid >= BLD) return;
  int d = (int)(tid & (DI - 1));
  int bl = (int)(tid >> 10);
  const float* dtr = xdbl + (size_t)bl * 40;
  const float* wr  = W + (size_t)d * 32;
  float acc = bias[d];
#pragma unroll
  for (int r = 0; r < 32; ++r) acc += dtr[r] * wr[r];
  dt[tid] = (acc > 20.f) ? acc : log1pf(__expf(acc));
}

// ---------------- chunked associative scan ---------------------------------
// h_t = dA_t h_{t-1} + dBx_t over L=4096, split into NCH chunks of CHL.
// Pass A: per (b,chunk,d): P[n] = prod dA, S[n] = chunk scan from h=0.
// Pass B: per (b,d): serially compose chunk states -> Hin per chunk.
// Pass C: per (b,chunk,d): replay chunk from Hin, emit gated y (f16).
__global__ void scan_passA(const float* __restrict__ dt, const float* __restrict__ xdbl,
                           const float* __restrict__ xc, const float* __restrict__ A_log,
                           float* __restrict__ Parr, float* __restrict__ Sarr) {
  int tid = blockIdx.x * blockDim.x + threadIdx.x;              // BB*NCH*DI
  if (tid >= BB * NCH * DI) return;
  int d = tid & (DI - 1), bc = tid >> 10, c = bc & (NCH - 1), b = bc >> 4;
  float A0 = -__expf(A_log[d * 4 + 0]), A1 = -__expf(A_log[d * 4 + 1]);
  float A2 = -__expf(A_log[d * 4 + 2]), A3 = -__expf(A_log[d * 4 + 3]);
  float P0 = 1.f, P1 = 1.f, P2 = 1.f, P3 = 1.f;
  float S0 = 0.f, S1 = 0.f, S2 = 0.f, S3 = 0.f;
  for (int tt = 0; tt < CHL; ++tt) {
    size_t bl = ((size_t)(b << 12)) + c * CHL + tt;
    float dtv = dt[bl * DI + d];
    float dx  = dtv * xc[bl * DI + d];
    const float* xd = xdbl + bl * 40;
    float e0 = __expf(dtv * A0), e1 = __expf(dtv * A1);
    float e2 = __expf(dtv * A2), e3 = __expf(dtv * A3);
    P0 *= e0; P1 *= e1; P2 *= e2; P3 *= e3;
    S0 = e0 * S0 + dx * xd[32]; S1 = e1 * S1 + dx * xd[33];
    S2 = e2 * S2 + dx * xd[34]; S3 = e3 * S3 + dx * xd[35];
  }
  int base = (b * NCH + c) * DI + d, st = BB * NCH * DI;
  Parr[0 * st + base] = P0; Parr[1 * st + base] = P1;
  Parr[2 * st + base] = P2; Parr[3 * st + base] = P3;
  Sarr[0 * st + base] = S0; Sarr[1 * st + base] = S1;
  Sarr[2 * st + base] = S2; Sarr[3 * st + base] = S3;
}

__global__ void scan_passB(const float* __restrict__ Parr, const float* __restrict__ Sarr,
                           float* __restrict__ Hin) {
  int tid = blockIdx.x * blockDim.x + threadIdx.x;              // BB*DI
  if (tid >= BB * DI) return;
  int d = tid & (DI - 1), b = tid >> 10;
  int st = BB * NCH * DI;
  float h0 = 0.f, h1 = 0.f, h2 = 0.f, h3 = 0.f;
  for (int c = 0; c < NCH; ++c) {
    int base = (b * NCH + c) * DI + d;
    Hin[0 * st + base] = h0; Hin[1 * st + base] = h1;
    Hin[2 * st + base] = h2; Hin[3 * st + base] = h3;
    h0 = Parr[0 * st + base] * h0 + Sarr[0 * st + base];
    h1 = Parr[1 * st + base] * h1 + Sarr[1 * st + base];
    h2 = Parr[2 * st + base] * h2 + Sarr[2 * st + base];
    h3 = Parr[3 * st + base] * h3 + Sarr[3 * st + base];
  }
}

__global__ void scan_passC(const float* __restrict__ dt, const float* __restrict__ xdbl,
                           const float* __restrict__ xc, const float* __restrict__ xz,
                           const float* __restrict__ A_log, const float* __restrict__ Dp,
                           const float* __restrict__ Hin, _Float16* __restrict__ y16) {
  int tid = blockIdx.x * blockDim.x + threadIdx.x;              // BB*NCH*DI
  if (tid >= BB * NCH * DI) return;
  int d = tid & (DI - 1), bc = tid >> 10, c = bc & (NCH - 1), b = bc >> 4;
  float A0 = -__expf(A_log[d * 4 + 0]), A1 = -__expf(A_log[d * 4 + 1]);
  float A2 = -__expf(A_log[d * 4 + 2]), A3 = -__expf(A_log[d * 4 + 3]);
  float Dd = Dp[d];
  int base = (b * NCH + c) * DI + d, st = BB * NCH * DI;
  float h0 = Hin[0 * st + base], h1 = Hin[1 * st + base];
  float h2 = Hin[2 * st + base], h3 = Hin[3 * st + base];
  for (int tt = 0; tt < CHL; ++tt) {
    size_t bl = ((size_t)(b << 12)) + c * CHL + tt;
    float dtv = dt[bl * DI + d];
    float xv  = xc[bl * DI + d];
    float zv  = xz[bl * 2048 + DI + d];
    const float* xd = xdbl + bl * 40;
    float dx = dtv * xv;
    h0 = __expf(dtv * A0) * h0 + dx * xd[32];
    h1 = __expf(dtv * A1) * h1 + dx * xd[33];
    h2 = __expf(dtv * A2) * h2 + dx * xd[34];
    h3 = __expf(dtv * A3) * h3 + dx * xd[35];
    float yv = h0 * xd[36] + h1 * xd[37] + h2 * xd[38] + h3 * xd[39];
    yv = (yv + xv * Dd) * (zv / (1.0f + __expf(-zv)));
    y16[bl * DI + d] = (_Float16)yv;
  }
}

// ---------------------------------------------------------------------------
extern "C" void kernel_launch(void* const* d_in, const int* in_sizes, int n_in,
                              void* d_out, int out_size, void* d_ws, size_t ws_size,
                              hipStream_t stream) {
  const float* hs       = (const float*)d_in[0];
  const int*   mask     = (const int*)d_in[1];
  const float* residual = (const float*)d_in[2];
  const float* norm_w   = (const float*)d_in[3];
  const float* conv_w   = (const float*)d_in[4];
  const float* conv_b   = (const float*)d_in[5];
  const float* ipw      = (const float*)d_in[6];
  const float* c1w      = (const float*)d_in[7];
  const float* c1b      = (const float*)d_in[8];
  const float* xpw      = (const float*)d_in[9];
  const float* dtw      = (const float*)d_in[10];
  const float* dtb      = (const float*)d_in[11];
  const float* A_log    = (const float*)d_in[12];
  const float* Dp       = (const float*)d_in[13];
  const float* opw      = (const float*)d_in[14];

  float* out_x   = (float*)d_out;            // (B,L,C)
  float* out_res = out_x + BLC;              // (B,L,C)

  // bump-allocate workspace (256B aligned)
  char* w = (char*)d_ws;
  size_t off = 0;
  auto alloc = [&](size_t bytes) { void* p = w + off; off = (off + bytes + 255) & ~(size_t)255; return p; };
  _Float16* nrm16   = (_Float16*)alloc((BLC + CC) * 2);   // +1 zeroed row (idx BL)
  _Float16* w9      = (_Float16*)alloc((size_t)9 * CC * CC * 2);
  _Float16* ipw16   = (_Float16*)alloc((size_t)2 * DI * CC * 2);
  _Float16* opw16   = (_Float16*)alloc((size_t)CC * DI * 2);
  _Float16* convo16 = (_Float16*)alloc(BLC * 2);
  _Float16* easy16  = (_Float16*)alloc(BLC * 2);
  int*      idx     = (int*)alloc((size_t)BB * LL * 4);
  float*    xzbuf   = (float*)alloc((size_t)BL * 2048 * 4);
  float*    xconv   = (float*)alloc(BLD * 4);
  float*    xdbl    = (float*)alloc((size_t)BL * 40 * 4);
  float*    dtbuf   = (float*)alloc(BLD * 4);
  float*    Parr    = (float*)alloc((size_t)4 * BB * NCH * DI * 4);
  float*    Sarr    = (float*)alloc((size_t)4 * BB * NCH * DI * 4);
  float*    Hin     = (float*)alloc((size_t)4 * BB * NCH * DI * 4);
  _Float16* y16     = (_Float16*)alloc(BLD * 2);
  (void)ws_size;

  // 1. residual + RMSNorm (also emits res output half)
  norm_kernel<<<BL, 256, 0, stream>>>(hs, residual, norm_w, out_res, nrm16);
  // 2. weight conversions / zero pad row / sort indices
  cvt_f16_kernel<<<(2 * DI * CC + 255) / 256, 256, 0, stream>>>(ipw, ipw16, 2 * DI * CC);
  cvt_f16_kernel<<<(CC * DI + 255) / 256, 256, 0, stream>>>(opw, opw16, CC * DI);
  repack_conv_w<<<(9 * CC * CC + 255) / 256, 256, 0, stream>>>(conv_w, w9);
  zero_f16_kernel<<<2, 256, 0, stream>>>(nrm16 + BLC, CC);
  build_idx<<<1, 32, 0, stream>>>(mask, idx);
  // 3. 3x3 conv (implicit WMMA GEMM, M=BL N=512 K=9*512), async-LDS B
  conv3x3_wmma<<<(BL / 128) * 8, 256, 0, stream>>>(nrm16, w9, conv_b, convo16);
  // 4. gather into easy order
  gather_rows<<<(int)(BLC / 8 + 255) / 256, 256, 0, stream>>>(convo16, idx, easy16);
  // 5. in_proj GEMM (M=16384, N=2048, K=512), async-LDS B
  gemm_f16_lds<<<(BL / 128) * (2048 / 64), 256, 0, stream>>>(easy16, ipw16, xzbuf, BL, 2048, CC);
  // 6. depthwise causal conv1d + SiLU
  dwconv1d_silu<<<(int)(BLD + 255) / 256, 256, 0, stream>>>(xzbuf, c1w, c1b, xconv);
  // 7. x_proj (N=40) and dt_proj (K=32) scalar
  xproj_kernel<<<(BL * 40 + 255) / 256, 256, 0, stream>>>(xconv, xpw, xdbl);
  dtproj_kernel<<<(int)(BLD + 255) / 256, 256, 0, stream>>>(xdbl, dtw, dtb, dtbuf);
  // 8. chunked selective scan (16x parallel) + D skip + z gate -> f16 Y
  scan_passA<<<(BB * NCH * DI + 255) / 256, 256, 0, stream>>>(dtbuf, xdbl, xconv, A_log, Parr, Sarr);
  scan_passB<<<(BB * DI + 255) / 256, 256, 0, stream>>>(Parr, Sarr, Hin);
  scan_passC<<<(BB * NCH * DI + 255) / 256, 256, 0, stream>>>(dtbuf, xdbl, xconv, xzbuf, A_log, Dp, Hin, y16);
  // 9. out_proj GEMM (async-LDS B) + scatter into original token order
  gemm_out_lds<<<(BL / 128) * 8, 256, 0, stream>>>(y16, opw16, idx, out_x);
}
